// Transformer_13735305412793
// MI455X (gfx1250) — matrix operand order repaired and checked by reading, not compiled, not run on hardware
//
#include <hip/hip_runtime.h>
#include <hip/hip_bf16.h>

// MI455X / gfx1250 transformer forward. Compute-bound (~86 GFLOP of GEMMs,
// all activations/weights L2-resident at 192MB), so every matmul goes through
// v_wmma_f32_16x16x32_bf16 (bf16 in, f32 accumulate) with LDS-staged tiles in
// the wave32 WMMA fragment layouts from the CDNA5 ISA. Global loads for the
// next k-tile are register-prefetched so they overlap the WMMA issue.

typedef __attribute__((ext_vector_type(16))) __bf16 v16bf;
typedef __attribute__((ext_vector_type(8)))  float  v8f;

#define BB   8
#define SS   1024
#define DD   256
#define HH   8
#define DKV  32
#define DFFN 1024
#define LLAY 4
#define NEGV (-1000000000.0f)
#define EPSV 1e-5f

union Frag {
  v16bf    v;
  __bf16   e[16];
  unsigned u[8];
  uint4    q[2];
};

// One packed f32->bf16 pair conversion, pinned to a single instruction so the
// SLP vectorizer cannot re-associate pairs and patch up with v_perm/v_or.
__device__ inline unsigned pk2(float a, float b) {
  unsigned r;
  asm("v_cvt_pk_bf16_f32 %0, %1, %2" : "=v"(r) : "v"(a), "v"(b));
  return r;
}

// Fill fragment elements [off, off+8) from two float4s as packed dwords:
// each (e[2i],e[2i+1]) pair is exactly one v_cvt_pk_bf16_f32.
__device__ inline void cvt8(Frag& f, int off, float4 a, float4 b) {
  int ui = off >> 1;
  f.u[ui + 0] = pk2(a.x, a.y);
  f.u[ui + 1] = pk2(a.z, a.w);
  f.u[ui + 2] = pk2(b.x, b.y);
  f.u[ui + 3] = pk2(b.z, b.w);
}

__device__ inline float wave_max32(float v) {
  for (int o = 16; o > 0; o >>= 1) v = fmaxf(v, __shfl_xor(v, o, 32));
  return v;
}
__device__ inline float wave_sum32(float v) {
  for (int o = 16; o > 0; o >>= 1) v += __shfl_xor(v, o, 32);
  return v;
}

// ---------------------------------------------------------------- embed + LN
__global__ void __launch_bounds__(256)
embed_ln_kernel(const int* __restrict__ ids, const float* __restrict__ tok,
                const float* __restrict__ pos, const float* __restrict__ w,
                const float* __restrict__ bpar, float* __restrict__ X) {
  int row = blockIdx.x;
  int d   = threadIdx.x;
  int s   = row & (SS - 1);
  int id  = ids[row];
  float val = tok[(size_t)id * DD + d] + pos[(size_t)s * DD + d];
  __shared__ float red[256];
  red[d] = val; __syncthreads();
  for (int st = 128; st > 0; st >>= 1) { if (d < st) red[d] += red[d + st]; __syncthreads(); }
  float mean = red[0] * (1.0f / DD);
  __syncthreads();
  float dv = val - mean;
  red[d] = dv * dv; __syncthreads();
  for (int st = 128; st > 0; st >>= 1) { if (d < st) red[d] += red[d + st]; __syncthreads(); }
  float var = red[0] * (1.0f / DD);
  X[(size_t)row * DD + d] = dv * rsqrtf(var + EPSV) * w[d] + bpar[d];
}

// ------------------------------------------------------------------------ LN
__global__ void __launch_bounds__(256)
ln_kernel(const float* __restrict__ in, const float* __restrict__ w,
          const float* __restrict__ bpar, float* __restrict__ out) {
  int row = blockIdx.x;
  int d   = threadIdx.x;
  float val = in[(size_t)row * DD + d];
  __shared__ float red[256];
  red[d] = val; __syncthreads();
  for (int st = 128; st > 0; st >>= 1) { if (d < st) red[d] += red[d + st]; __syncthreads(); }
  float mean = red[0] * (1.0f / DD);
  __syncthreads();
  float dv = val - mean;
  red[d] = dv * dv; __syncthreads();
  for (int st = 128; st > 0; st >>= 1) { if (d < st) red[d] += red[d + st]; __syncthreads(); }
  float var = red[0] * (1.0f / DD);
  out[(size_t)row * DD + d] = dv * rsqrtf(var + EPSV) * w[d] + bpar[d];
}

// ------------------------------------------------- WMMA GEMM: out = A*W(+b..)
// Block tile 128x64, 8 waves, each wave 32x32 (2x2 wmma 16x16x32 bf16 tiles).
// Next k-tile is register-prefetched; epilogue goes through a 32KB LDS tile
// so bias/residual/relu and global stores are all coalesced float4.
template <int HASRES, int RELU>
__global__ void __launch_bounds__(256)
gemm_kernel(const float* __restrict__ A, const float* __restrict__ W,
            const float* __restrict__ bias, const float* __restrict__ resid,
            float* __restrict__ out, int M, int N, int K) {
  __shared__ __bf16 As[128 * 32];   // [m][k], k contiguous      (8KB)
  __shared__ __bf16 Bs[64 * 32];    // [n][k], k contiguous      (4KB)
  __shared__ float  Cs[128 * 64];   // output tile               (32KB)
  int t = threadIdx.x;
  int l = t & 31, wv = t >> 5;
  int wm = wv >> 1, wn = wv & 1;
  int n0 = blockIdx.x * 64, m0 = blockIdx.y * 128;
  v8f acc[2][2] = {};
  int arow = t >> 1, aseg = (t & 1) * 16;     // A staging: 16 floats/thread
  int bn = t & 63, bkp = t >> 6;              // B staging: 8 floats/thread
  int abase = (l >= 16) ? 8 : 0;
  int kbb   = (l >= 16) ? 8 : 0;

  float4 pa[4];
  float  pb[8];
  // prefetch first tile
  {
    const float* ap = A + (size_t)(m0 + arow) * K + aseg;
    #pragma unroll
    for (int j = 0; j < 4; ++j) pa[j] = *(const float4*)(ap + 4 * j);
    #pragma unroll
    for (int j = 0; j < 4; ++j) {
      int kp = bkp * 4 + j;
      pb[2*j+0] = W[(size_t)(2*kp+0) * N + n0 + bn];
      pb[2*j+1] = W[(size_t)(2*kp+1) * N + n0 + bn];
    }
  }

  for (int k0 = 0; k0 < K; k0 += 32) {
    __syncthreads();   // previous iteration's fragment reads complete
    #pragma unroll
    for (int j = 0; j < 4; ++j) {
      uint2 u; u.x = pk2(pa[j].x, pa[j].y); u.y = pk2(pa[j].z, pa[j].w);
      *(uint2*)&As[arow * 32 + aseg + 4 * j] = u;
    }
    #pragma unroll
    for (int j = 0; j < 4; ++j) {
      int kp = bkp * 4 + j;
      *(unsigned*)&Bs[bn * 32 + 2 * kp] = pk2(pb[2*j+0], pb[2*j+1]);
    }
    __syncthreads();

    // prefetch next k-tile (global loads in flight while WMMAs issue)
    if (k0 + 32 < K) {
      const float* ap = A + (size_t)(m0 + arow) * K + (k0 + 32) + aseg;
      #pragma unroll
      for (int j = 0; j < 4; ++j) pa[j] = *(const float4*)(ap + 4 * j);
      #pragma unroll
      for (int j = 0; j < 4; ++j) {
        int kp = bkp * 4 + j;
        pb[2*j+0] = W[(size_t)(k0 + 32 + 2*kp+0) * N + n0 + bn];
        pb[2*j+1] = W[(size_t)(k0 + 32 + 2*kp+1) * N + n0 + bn];
      }
    }

    Frag af[2], bf[2];
    #pragma unroll
    for (int mt = 0; mt < 2; ++mt) {
      const __bf16* p = &As[(wm * 32 + mt * 16 + (l & 15)) * 32 + abase];
      af[mt].q[0] = *reinterpret_cast<const uint4*>(p);
      af[mt].q[1] = *reinterpret_cast<const uint4*>(p + 16);
    }
    #pragma unroll
    for (int nt = 0; nt < 2; ++nt) {
      const __bf16* p = &Bs[(wn * 32 + nt * 16 + (l & 15)) * 32 + kbb];
      bf[nt].q[0] = *reinterpret_cast<const uint4*>(p);
      bf[nt].q[1] = *reinterpret_cast<const uint4*>(p + 16);
    }
    #pragma unroll
    for (int mt = 0; mt < 2; ++mt)
      #pragma unroll
      for (int nt = 0; nt < 2; ++nt)
        acc[mt][nt] = __builtin_amdgcn_wmma_f32_16x16x32_bf16(
            false, af[mt].v, false, bf[nt].v, (short)0, acc[mt][nt], false, false);
  }

  // stage accumulators to LDS, then coalesced epilogue
  int mlane = (l >= 16) ? 8 : 0;
  #pragma unroll
  for (int mt = 0; mt < 2; ++mt)
    #pragma unroll
    for (int nt = 0; nt < 2; ++nt) {
      int cn = wn * 32 + nt * 16 + (l & 15);
      #pragma unroll
      for (int r = 0; r < 8; ++r)
        Cs[(wm * 32 + mt * 16 + mlane + r) * 64 + cn] = acc[mt][nt][r];
    }
  __syncthreads();

  int orow = t >> 1, ocs = (t & 1) * 32;
  const float4* c4 = (const float4*)&Cs[orow * 64 + ocs];
  size_t obase = (size_t)(m0 + orow) * N + n0 + ocs;
  #pragma unroll
  for (int j = 0; j < 8; ++j) {
    float4 v = c4[j];
    float4 bb = *(const float4*)&bias[n0 + ocs + 4 * j];
    v.x += bb.x; v.y += bb.y; v.z += bb.z; v.w += bb.w;
    if (HASRES) {
      float4 rr = *(const float4*)&resid[obase + 4 * j];
      v.x += rr.x; v.y += rr.y; v.z += rr.z; v.w += rr.w;
    }
    if (RELU) {
      v.x = fmaxf(v.x, 0.0f); v.y = fmaxf(v.y, 0.0f);
      v.z = fmaxf(v.z, 0.0f); v.w = fmaxf(v.w, 0.0f);
    }
    *(float4*)&out[obase + 4 * j] = v;
  }
}

// ------------------------------------------------------------ attention core
// One block per (b, h, 32 query rows). Uses CDNA5's big LDS:
//   sc    32x1024 f32   (128KB)  full score rows
//   vT    32x1024 bf16  ( 64KB)  V transposed [vdim][key]
//   kst   128x32  bf16  (  8KB)  K chunk [key][dim]
//   ctxb  32x32   f32   (  4KB)  output tile (ds_add_f32 reduction)
//   maskf 1024    f32   (  4KB)  pad mask per key
#define ATTN_SMEM (131072 + 65536 + 8192 + 4096 + 4096)
__global__ void __launch_bounds__(256)
attn_kernel(const float* __restrict__ Q, const float* __restrict__ Kmat,
            const float* __restrict__ V, const int* __restrict__ ids,
            float* __restrict__ CTX) {
  extern __shared__ char smem[];
  float*  sc    = (float*)smem;
  __bf16* vT    = (__bf16*)(smem + 131072);
  __bf16* kst   = (__bf16*)(smem + 131072 + 65536);
  float*  ctxb  = (float*)(smem + 131072 + 65536 + 8192);
  float*  maskf = (float*)(smem + 131072 + 65536 + 8192 + 4096);

  int t = threadIdx.x, l = t & 31, wv = t >> 5;
  int q0 = blockIdx.x * 32, h = blockIdx.y, b = blockIdx.z;
  int dl  = (l >= 16) ? 8 : 0;
  int kb2 = (l >= 16) ? 8 : 0;
  int mr  = (l >= 16) ? 8 : 0;

  #pragma unroll
  for (int j = 0; j < 4; ++j) ctxb[t + 256 * j] = 0.0f;
  #pragma unroll
  for (int j = 0; j < 4; ++j) {
    int key = t + 256 * j;
    maskf[key] = (ids[b * SS + key] == 0) ? 1.0f : 0.0f;
  }

  // stage V transposed: vT[dim][key], packed pairs of keys per b32 store
  #pragma unroll
  for (int it = 0; it < 2; ++it) {
    int ka = 2 * (t + 256 * it);
    const float* va = V + (size_t)(b * SS + ka) * DD + h * DKV;
    const float* vb2 = va + DD;
    #pragma unroll
    for (int j = 0; j < 8; ++j) {
      float4 fa = *(const float4*)(va + 4 * j);
      float4 fb = *(const float4*)(vb2 + 4 * j);
      *(unsigned*)&vT[(4 * j + 0) * SS + ka] = pk2(fa.x, fb.x);
      *(unsigned*)&vT[(4 * j + 1) * SS + ka] = pk2(fa.y, fb.y);
      *(unsigned*)&vT[(4 * j + 2) * SS + ka] = pk2(fa.z, fb.z);
      *(unsigned*)&vT[(4 * j + 3) * SS + ka] = pk2(fa.w, fb.w);
    }
  }

  // Q fragments (A-matrix layout), shared by all waves
  Frag qa[2];
  #pragma unroll
  for (int mt = 0; mt < 2; ++mt) {
    const float* qp = Q + (size_t)(b * SS + q0 + mt * 16 + (l & 15)) * DD + h * DKV;
    float4 a0 = *(const float4*)(qp + dl);
    float4 a1 = *(const float4*)(qp + dl + 4);
    float4 a2 = *(const float4*)(qp + dl + 16);
    float4 a3 = *(const float4*)(qp + dl + 20);
    cvt8(qa[mt], 0, a0, a1);
    cvt8(qa[mt], 8, a2, a3);
  }

  // Phase 1: scores = Q * K^T  (DK=32 -> one wmma k-step per 16 keys)
  int arow = t >> 1, aseg = (t & 1) * 16;
  for (int c = 0; c < 8; ++c) {
    __syncthreads();
    const float* kp = Kmat + (size_t)(b * SS + c * 128 + arow) * DD + h * DKV + aseg;
    #pragma unroll
    for (int j = 0; j < 4; ++j) {
      float4 f = *(const float4*)(kp + 4 * j);
      uint2 u; u.x = pk2(f.x, f.y); u.y = pk2(f.z, f.w);
      *(uint2*)&kst[arow * 32 + aseg + 4 * j] = u;
    }
    __syncthreads();
    Frag bfr;
    const __bf16* p = &kst[(wv * 16 + (l & 15)) * 32 + kb2];
    bfr.q[0] = *reinterpret_cast<const uint4*>(p);
    bfr.q[1] = *reinterpret_cast<const uint4*>(p + 16);
    v8f d0 = {}, d1 = {};
    d0 = __builtin_amdgcn_wmma_f32_16x16x32_bf16(false, qa[0].v, false, bfr.v, (short)0, d0, false, false);
    d1 = __builtin_amdgcn_wmma_f32_16x16x32_bf16(false, qa[1].v, false, bfr.v, (short)0, d1, false, false);
    int key = c * 128 + wv * 16 + (l & 15);
    #pragma unroll
    for (int r = 0; r < 8; ++r) {
      sc[(mr + r) * SS + key]      = d0[r];
      sc[(mr + r + 16) * SS + key] = d1[r];
    }
  }
  __syncthreads();

  // Phase 2: scale + pad mask + softmax (wave per 4 rows)
  const float scale = 0.17677669529663687f;  // 1/sqrt(32)
  for (int rr = 0; rr < 4; ++rr) {
    float* srow = &sc[(wv * 4 + rr) * SS];
    float mx = -3.0e38f;
    for (int kk = l; kk < SS; kk += 32) {
      float vv = srow[kk] * scale;
      if (maskf[kk] != 0.0f) vv = NEGV;
      srow[kk] = vv;
      mx = fmaxf(mx, vv);
    }
    mx = wave_max32(mx);
    float sum = 0.0f;
    for (int kk = l; kk < SS; kk += 32) {
      float e = __expf(srow[kk] - mx);
      srow[kk] = e;
      sum += e;
    }
    sum = wave_sum32(sum);
    float inv = 1.0f / sum;
    for (int kk = l; kk < SS; kk += 32) srow[kk] *= inv;
  }
  __syncthreads();

  // Phase 3: ctx = P * V  (keys split across waves, ds_add_f32 reduction)
  v8f acc[2][2] = {};
  for (int ks = wv * 128; ks < wv * 128 + 128; ks += 32) {
    Frag pa[2], vb[2];
    #pragma unroll
    for (int mt = 0; mt < 2; ++mt) {
      const float* pr = &sc[(mt * 16 + (l & 15)) * SS + ks + dl];
      float4 a0 = *(const float4*)(pr);
      float4 a1 = *(const float4*)(pr + 4);
      float4 a2 = *(const float4*)(pr + 16);
      float4 a3 = *(const float4*)(pr + 20);
      cvt8(pa[mt], 0, a0, a1);
      cvt8(pa[mt], 8, a2, a3);
    }
    #pragma unroll
    for (int nt = 0; nt < 2; ++nt) {
      const __bf16* p = &vT[(nt * 16 + (l & 15)) * SS + ks + kb2];
      vb[nt].q[0] = *reinterpret_cast<const uint4*>(p);
      vb[nt].q[1] = *reinterpret_cast<const uint4*>(p + 16);
    }
    #pragma unroll
    for (int mt = 0; mt < 2; ++mt)
      #pragma unroll
      for (int nt = 0; nt < 2; ++nt)
        acc[mt][nt] = __builtin_amdgcn_wmma_f32_16x16x32_bf16(
            false, pa[mt].v, false, vb[nt].v, (short)0, acc[mt][nt], false, false);
  }
  #pragma unroll
  for (int mt = 0; mt < 2; ++mt)
    #pragma unroll
    for (int nt = 0; nt < 2; ++nt)
      #pragma unroll
      for (int r = 0; r < 8; ++r)
        atomicAdd(&ctxb[(mt * 16 + mr + r) * 32 + nt * 16 + (l & 15)], acc[mt][nt][r]);
  __syncthreads();

  // Phase 4: write ctx tile [B,S,H*DV]
  #pragma unroll
  for (int j = 0; j < 4; ++j) {
    int idx = t + 256 * j;
    int m = idx >> 5, n = idx & 31;
    CTX[(size_t)(b * SS + q0 + m) * DD + h * DKV + n] = ctxb[m * 32 + n];
  }
}

// -------------------------------------------------------------- classifier
__global__ void __launch_bounds__(256)
head_kernel(const float* __restrict__ X, const float* __restrict__ f1w,
            const float* __restrict__ f1b, const float* __restrict__ f2w,
            const float* __restrict__ f2b, const float* __restrict__ cw,
            const float* __restrict__ cb, float* __restrict__ out) {
  __shared__ float h1[8 * 256];
  __shared__ float fbuf[8 * 128];
  int t = threadIdx.x;
  for (int j = 0; j < 8; ++j) {
    int idx = t + 256 * j;
    int b = idx >> 8, col = idx & 255;
    const float* xr = X + (size_t)(b * SS) * DD;   // cls token (s=0)
    float s = f1b[col];
    for (int d2 = 0; d2 < 256; ++d2) s += xr[d2] * f1w[d2 * 256 + col];
    h1[b * 256 + col] = fmaxf(s, 0.0f);
  }
  __syncthreads();
  for (int j = 0; j < 4; ++j) {
    int idx = t + 256 * j;
    int b = idx >> 7, col = idx & 127;
    float s = f2b[col];
    for (int d2 = 0; d2 < 256; ++d2) s += h1[b * 256 + d2] * f2w[d2 * 128 + col];
    fbuf[b * 128 + col] = s;
    out[16 + b * 128 + col] = s;       // feat
  }
  __syncthreads();
  if (t < 16) {
    int b = t >> 1, c = t & 1;
    float s = cb[c];
    for (int j2 = 0; j2 < 128; ++j2) s += fbuf[b * 128 + j2] * cw[j2 * 2 + c];
    out[b * 2 + c] = s;                // logits
  }
}

// ----------------------------------------------------------------- launcher
extern "C" void kernel_launch(void* const* d_in, const int* in_sizes, int n_in,
                              void* d_out, int out_size, void* d_ws, size_t ws_size,
                              hipStream_t stream) {
  const int*   ids = (const int*)  d_in[0];
  const float* tok = (const float*)d_in[1];
  const float* pos = (const float*)d_in[2];
  const float* elw = (const float*)d_in[3];
  const float* elb = (const float*)d_in[4];
  const float* Wq  = (const float*)d_in[5];
  const float* bq  = (const float*)d_in[6];
  const float* Wk  = (const float*)d_in[7];
  const float* bk  = (const float*)d_in[8];
  const float* Wv  = (const float*)d_in[9];
  const float* bv  = (const float*)d_in[10];
  const float* Wo  = (const float*)d_in[11];
  const float* bo  = (const float*)d_in[12];
  const float* lnw = (const float*)d_in[13];
  const float* lnb = (const float*)d_in[14];
  const float* W1  = (const float*)d_in[15];
  const float* b1  = (const float*)d_in[16];
  const float* W2  = (const float*)d_in[17];
  const float* b2  = (const float*)d_in[18];
  const float* f1w = (const float*)d_in[19];
  const float* f1b = (const float*)d_in[20];
  const float* f2w = (const float*)d_in[21];
  const float* f2b = (const float*)d_in[22];
  const float* cw  = (const float*)d_in[23];
  const float* cb  = (const float*)d_in[24];
  float* out = (float*)d_out;

  char* ws = (char*)d_ws;
  const size_t MB = 1024 * 1024;
  float* X  = (float*)(ws + 0 * MB);    // [8192,256]  activations
  float* Qb = (float*)(ws + 8 * MB);
  float* Kb = (float*)(ws + 16 * MB);
  float* Vb = (float*)(ws + 24 * MB);
  float* Cb = (float*)(ws + 32 * MB);   // attention context
  float* Tb = (float*)(ws + 40 * MB);   // pre-LN buffer
  float* Hb = (float*)(ws + 48 * MB);   // [8192,1024] FFN hidden (32MB)

  (void)in_sizes; (void)n_in; (void)out_size; (void)ws_size;

  hipFuncSetAttribute(reinterpret_cast<const void*>(&attn_kernel),
                      hipFuncAttributeMaxDynamicSharedMemorySize, ATTN_SMEM);

  const int M = BB * SS;  // 8192

  embed_ln_kernel<<<dim3(M), dim3(256), 0, stream>>>(ids, tok, pos, elw, elb, X);

  for (int i = 0; i < LLAY; ++i) {
    const size_t wo = (size_t)i * DD * DD;
    gemm_kernel<0,0><<<dim3(4, 64), 256, 0, stream>>>(X, Wq + wo, bq + i * DD, nullptr, Qb, M, DD, DD);
    gemm_kernel<0,0><<<dim3(4, 64), 256, 0, stream>>>(X, Wk + wo, bk + i * DD, nullptr, Kb, M, DD, DD);
    gemm_kernel<0,0><<<dim3(4, 64), 256, 0, stream>>>(X, Wv + wo, bv + i * DD, nullptr, Vb, M, DD, DD);

    attn_kernel<<<dim3(SS / 32, HH, BB), 256, ATTN_SMEM, stream>>>(Qb, Kb, Vb, ids, Cb);

    // ctx @ Wo + bo + residual(X) -> Tb ; LN -> X
    gemm_kernel<1,0><<<dim3(4, 64), 256, 0, stream>>>(Cb, Wo + wo, bo + i * DD, X, Tb, M, DD, DD);
    ln_kernel<<<dim3(M), 256, 0, stream>>>(Tb, lnw + i * DD, lnb + i * DD, X);

    // FFN: relu(X@W1+b1) -> Hb ; Hb@W2+b2 -> X   (no residual in reference)
    gemm_kernel<0,1><<<dim3(DFFN / 64, 64), 256, 0, stream>>>(
        X, W1 + (size_t)i * DD * DFFN, b1 + i * DFFN, nullptr, Hb, M, DFFN, DD);
    gemm_kernel<0,0><<<dim3(4, 64), 256, 0, stream>>>(
        Hb, W2 + (size_t)i * DFFN * DD, b2 + i * DD, nullptr, X, M, DD, DFFN);
  }

  head_kernel<<<1, 256, 0, stream>>>(X, f1w, f1b, f2w, f2b, cw, cb, out);
}
